// LocalTransformerEncoderLayer_10299331575846
// MI455X (gfx1250) — matrix operand order repaired and checked by reading, not compiled, not run on hardware
//
#include <hip/hip_runtime.h>

// ---------------- problem constants ----------------
#define B_   4
#define S_   2048
#define E_   512
#define H_   8
#define DH_  64
#define F_   2048
#define WIN_ 128
#define M_   (B_*S_)   // 8192 rows total

// ---------------- CDNA5 async global->LDS copy availability ----------------
#if defined(__has_builtin)
#  if __has_builtin(__builtin_amdgcn_global_load_async_to_lds_b128) && \
      __has_builtin(__builtin_amdgcn_s_wait_asynccnt)
#    define USE_ASYNC_LDS 1
#  endif
#endif
#ifndef USE_ASYNC_LDS
#  define USE_ASYNC_LDS 0
#endif

// The builtin's params are int4-typed address-space pointers:
//   (int4 addrspace(1)* gsrc, int4 addrspace(3)* ldst, imm offset, imm cpol)
typedef int int4v __attribute__((ext_vector_type(4)));
typedef __attribute__((address_space(1))) int4v g_int4;
typedef __attribute__((address_space(3))) int4v l_int4;

// ---------------- WMMA types (CDNA5 / gfx1250, wave32) ----------------
typedef _Float16 half16 __attribute__((ext_vector_type(16)));
typedef float    v8f    __attribute__((ext_vector_type(8)));

union Frag { half16 h; uint4 u[2]; };

__device__ __forceinline__ v8f wmma_f16(half16 a, half16 b, v8f c) {
  // D = A(16x32 f16) * B(32x16 f16) + C(16x16 f32)
  return __builtin_amdgcn_wmma_f32_16x16x32_f16(
      /*neg_a=*/false, a, /*neg_b=*/false, b,
      /*c_mod=*/(short)0, c, /*reuse_a=*/false, /*reuse_b=*/false);
}

// ---------------- fp32 -> fp16 convert ----------------
__global__ void f32_to_f16_kernel(const float* __restrict__ in,
                                  _Float16* __restrict__ out, int n) {
  int i = blockIdx.x * blockDim.x + threadIdx.x;
  if (i < n) out[i] = (_Float16)in[i];
}

// ---------------- WMMA GEMM: C[M,N] = act(A[M,K] @ W[N,K]^T + bias) (+resid) ----------------
// A,W f16 row-major. Block tile 128x128, K-step 32, 8 waves (2x4 wmma tiles/wave).
// Pipelined: tile i+1 fetch issued before tile i compute (async LDS copy when available).
__global__ __launch_bounds__(256)
void wmma_gemm_kernel(const _Float16* __restrict__ A, const _Float16* __restrict__ Wm,
                      const float* __restrict__ bias, const float* __restrict__ resid,
                      float* __restrict__ Cf, _Float16* __restrict__ Ch,
                      int M, int N, int K, int do_relu)
{
  const int t    = threadIdx.x;
  const int lane = t & 31, wave = t >> 5;
  const int wm   = wave >> 1;      // 0..3 (32-row strip)
  const int wn   = wave & 1;       // 0..1 (64-col strip)
  const int lo   = lane & 15;      // row (A-frag) / col (B-frag, C tile)
  const int g    = lane >> 4;      // lane-group
  const int bm   = blockIdx.y * 128, bn = blockIdx.x * 128;

  const int ldr = t >> 1;              // 0..127: tile row loaded by this thread
  const int ldc = (t & 1) * 16;        // halves 0..15 or 16..31

  v8f zero = {0.f,0.f,0.f,0.f,0.f,0.f,0.f,0.f};
  v8f acc[2][4];
  #pragma unroll
  for (int i = 0; i < 2; ++i)
    #pragma unroll
    for (int j = 0; j < 4; ++j) acc[i][j] = zero;

  const _Float16* gArow = A  + (size_t)(bm + ldr) * K + ldc;
  const _Float16* gBrow = Wm + (size_t)(bn + ldr) * K + ldc;

#if USE_ASYNC_LDS
  __shared__ __align__(16) _Float16 sA[2][128 * 32];
  __shared__ __align__(16) _Float16 sB[2][128 * 32];

  auto issue_tile = [&](int k0, int buf) {
    __builtin_amdgcn_global_load_async_to_lds_b128(
        (g_int4*)(gArow + k0),     (l_int4*)&sA[buf][ldr * 32 + ldc],     0, 0);
    __builtin_amdgcn_global_load_async_to_lds_b128(
        (g_int4*)(gArow + k0 + 8), (l_int4*)&sA[buf][ldr * 32 + ldc + 8], 0, 0);
    __builtin_amdgcn_global_load_async_to_lds_b128(
        (g_int4*)(gBrow + k0),     (l_int4*)&sB[buf][ldr * 32 + ldc],     0, 0);
    __builtin_amdgcn_global_load_async_to_lds_b128(
        (g_int4*)(gBrow + k0 + 8), (l_int4*)&sB[buf][ldr * 32 + ldc + 8], 0, 0);
  };
#else
  __shared__ __align__(16) _Float16 sA1[128 * 32];
  __shared__ __align__(16) _Float16 sB1[128 * 32];
#endif

  auto compute_tile = [&](const _Float16* tA, const _Float16* tB) {
    Frag af[2], bf[4];
    #pragma unroll
    for (int mt = 0; mt < 2; ++mt) {       // A-layout: e<8 -> K=g*8+e ; e>=8 -> K=16+g*8+(e-8)
      int r = wm * 32 + mt * 16 + lo;
      af[mt].u[0] = *(const uint4*)&tA[r * 32 + g * 8];
      af[mt].u[1] = *(const uint4*)&tA[r * 32 + 16 + g * 8];
    }
    #pragma unroll
    for (int nt = 0; nt < 4; ++nt) {       // B-layout: e<8 -> K=g*8+e ; e>=8 -> K=16+g*8+(e-8)
      int r = wn * 64 + nt * 16 + lo;
      bf[nt].u[0] = *(const uint4*)&tB[r * 32 + g * 8];
      bf[nt].u[1] = *(const uint4*)&tB[r * 32 + 16 + g * 8];
    }
    #pragma unroll
    for (int mt = 0; mt < 2; ++mt)
      #pragma unroll
      for (int nt = 0; nt < 4; ++nt)
        acc[mt][nt] = wmma_f16(af[mt].h, bf[nt].h, acc[mt][nt]);
  };

#if USE_ASYNC_LDS
  const int nIter = K >> 5;
  issue_tile(0, 0);                               // prologue
  for (int i = 0; i < nIter; ++i) {
    const int buf = i & 1;
    if (i + 1 < nIter) {
      issue_tile((i + 1) << 5, buf ^ 1);          // overlap next fetch with compute
      __builtin_amdgcn_s_wait_asynccnt(4);        // my tile-i copies complete
    } else {
      __builtin_amdgcn_s_wait_asynccnt(0);
    }
    __syncthreads();                              // everyone's copies visible
    compute_tile(&sA[buf][0], &sB[buf][0]);
    __syncthreads();                              // reads done before buf is rewritten
  }
#else
  // register double-buffer: fetch tile i+1 into regs while computing tile i
  uint4 ra0 = *(const uint4*)gArow,       ra1 = *(const uint4*)(gArow + 8);
  uint4 rb0 = *(const uint4*)gBrow,       rb1 = *(const uint4*)(gBrow + 8);
  for (int k0 = 0; k0 < K; k0 += 32) {
    __syncthreads();                              // protect prior tile reads
    *(uint4*)&sA1[ldr * 32 + ldc]     = ra0;
    *(uint4*)&sA1[ldr * 32 + ldc + 8] = ra1;
    *(uint4*)&sB1[ldr * 32 + ldc]     = rb0;
    *(uint4*)&sB1[ldr * 32 + ldc + 8] = rb1;
    __syncthreads();
    if (k0 + 32 < K) {
      const _Float16* gA = gArow + k0 + 32;
      const _Float16* gB = gBrow + k0 + 32;
      if (k0 + 64 < K) {                          // L2 hint two slabs ahead
        __builtin_prefetch(gA + 32, 0, 1);
        __builtin_prefetch(gB + 32, 0, 1);
      }
      ra0 = *(const uint4*)gA;  ra1 = *(const uint4*)(gA + 8);
      rb0 = *(const uint4*)gB;  rb1 = *(const uint4*)(gB + 8);
    }
    compute_tile(sA1, sB1);                       // loads above retire under WMMAs
  }
#endif

  // C/D layout: lane-group g -> rows g*8..g*8+7 (per VGPR r), col = lo
  const int mb = g * 8;
  #pragma unroll
  for (int mt = 0; mt < 2; ++mt) {
    #pragma unroll
    for (int nt = 0; nt < 4; ++nt) {
      int colN = bn + wn * 64 + nt * 16 + lo;
      float bv = bias ? bias[colN] : 0.f;
      #pragma unroll
      for (int r = 0; r < 8; ++r) {
        int rowM = bm + wm * 32 + mt * 16 + mb + r;
        float v = acc[mt][nt][r] + bv;
        if (do_relu) v = fmaxf(v, 0.f);
        size_t idx = (size_t)rowM * N + colN;
        if (resid) v += resid[idx];
        if (Cf) Cf[idx] = v;
        if (Ch) Ch[idx] = (_Float16)v;
      }
    }
  }
}

// ---------------- qkv [M,3E] -> Q,K [B,H,S,64], Vt [B,H,64,S] ----------------
__global__ void qkv_reshape_kernel(const _Float16* __restrict__ qkv,
                                   _Float16* __restrict__ Q,
                                   _Float16* __restrict__ Kt,
                                   _Float16* __restrict__ Vt)
{
  int idx = blockIdx.x * blockDim.x + threadIdx.x;   // [0, M_*E_)
  int rs = idx >> 9;            // b*S + s
  int c  = idx & 511;
  int b  = rs >> 11;
  int s  = rs & (S_ - 1);
  int h  = c >> 6, d = c & 63;
  size_t src = (size_t)rs * (3 * E_);
  size_t bh  = (size_t)(b * H_ + h);
  Q [(bh * S_ + s) * DH_ + d] = qkv[src + c];
  Kt[(bh * S_ + s) * DH_ + d] = qkv[src + E_ + c];
  Vt[(bh * DH_ + d) * S_ + s] = qkv[src + 2 * E_ + c];
}

// ---------------- banded flash attention, one wave per 16-query block ----------------
__global__ __launch_bounds__(32)
void attn_kernel(const _Float16* __restrict__ Qh, const _Float16* __restrict__ Kh,
                 const _Float16* __restrict__ Vt, _Float16* __restrict__ Out)
{
  __shared__ __align__(16) _Float16 sP[16 * 32];   // P tile staging (C-layout -> A-layout)
  const int lane = threadIdx.x & 31;
  const int lo = lane & 15, g = lane >> 4;
  const int q0 = blockIdx.x * 16;
  const int h  = blockIdx.y;
  const int b  = blockIdx.z;

  const _Float16* Qp = Qh + (((size_t)(b * H_ + h)) * S_ + q0) * DH_;
  const _Float16* Kp = Kh + (((size_t)(b * H_ + h)) * S_) * DH_;
  const _Float16* Vp = Vt + (((size_t)(b * H_ + h)) * DH_) * S_;

  Frag qa[2];                                       // Q 16x64 as two 16x32 A-frags
  #pragma unroll
  for (int c = 0; c < 2; ++c) {
    qa[c].u[0] = *(const uint4*)&Qp[lo * DH_ + c * 32 + g * 8];
    qa[c].u[1] = *(const uint4*)&Qp[lo * DH_ + c * 32 + 16 + g * 8];
  }

  v8f zero = {0.f,0.f,0.f,0.f,0.f,0.f,0.f,0.f};
  v8f o[4];
  #pragma unroll
  for (int i = 0; i < 4; ++i) o[i] = zero;
  float mi[8], li[8];
  #pragma unroll
  for (int r = 0; r < 8; ++r) { mi[r] = -1e30f; li[r] = 0.f; }

  int klo = q0 - WIN_; if (klo < 0) klo = 0; klo &= ~31;
  int khi = q0 + 15 + WIN_; if (khi > S_ - 1) khi = S_ - 1;

  for (int kc = klo; kc <= khi; kc += 32) {         // kc+31 <= S-1 always (S % 32 == 0)
    // --- scores: S = Q @ K^T, 16x32 per chunk ---
    v8f sc[2];
    #pragma unroll
    for (int nt = 0; nt < 2; ++nt) {
      Frag kb[2];
      int krow = kc + nt * 16 + lo;
      #pragma unroll
      for (int c = 0; c < 2; ++c) {                 // B-frag: elements = Dh (g*8+e ; 16+g*8+(e-8))
        kb[c].u[0] = *(const uint4*)&Kp[(size_t)krow * DH_ + c * 32 + g * 8];
        kb[c].u[1] = *(const uint4*)&Kp[(size_t)krow * DH_ + c * 32 + 16 + g * 8];
      }
      v8f a = zero;
      a = wmma_f16(qa[0].h, kb[0].h, a);
      a = wmma_f16(qa[1].h, kb[1].h, a);
      sc[nt] = a;
    }
    // --- band mask + online softmax ---
    float mnew[8], alpha[8];
    #pragma unroll
    for (int r = 0; r < 8; ++r) {
      int q = q0 + g * 8 + r;
      float rowm = -1e30f;
      #pragma unroll
      for (int nt = 0; nt < 2; ++nt) {
        int key = kc + nt * 16 + lo;
        float v = sc[nt][r] * 0.125f;               // 1/sqrt(64)
        int dq = q - key;
        if (key >= S_ || dq > WIN_ || dq < -WIN_) v = -3.0e38f;
        sc[nt][r] = v;
        rowm = fmaxf(rowm, v);
      }
      #pragma unroll
      for (int off = 1; off < 16; off <<= 1)
        rowm = fmaxf(rowm, __shfl_xor(rowm, off, 32));
      mnew[r]  = fmaxf(mi[r], rowm);
      alpha[r] = exp2f((mi[r] - mnew[r]) * 1.44269504f);
      mi[r]    = mnew[r];
    }
    #pragma unroll
    for (int r = 0; r < 8; ++r) {
      float rs = 0.f;
      #pragma unroll
      for (int nt = 0; nt < 2; ++nt) {
        float p = exp2f((sc[nt][r] - mnew[r]) * 1.44269504f);
        sc[nt][r] = p;
        rs += p;
      }
      #pragma unroll
      for (int off = 1; off < 16; off <<= 1)
        rs += __shfl_xor(rs, off, 32);
      li[r] = li[r] * alpha[r] + rs;
      #pragma unroll
      for (int d = 0; d < 4; ++d) o[d][r] *= alpha[r];
    }
    // --- P (C-layout) -> LDS -> A-layout frag ---
    __syncthreads();
    #pragma unroll
    for (int r = 0; r < 8; ++r)
      #pragma unroll
      for (int nt = 0; nt < 2; ++nt)
        sP[(g * 8 + r) * 32 + nt * 16 + lo] = (_Float16)sc[nt][r];
    __syncthreads();
    Frag pf;
    pf.u[0] = *(const uint4*)&sP[lo * 32 + g * 8];
    pf.u[1] = *(const uint4*)&sP[lo * 32 + 16 + g * 8];
    // --- O += P @ V (V^T rows are contiguous in key) ---
    #pragma unroll
    for (int d = 0; d < 4; ++d) {
      Frag vf;
      int drow = d * 16 + lo;
      vf.u[0] = *(const uint4*)&Vp[(size_t)drow * S_ + kc + g * 8];
      vf.u[1] = *(const uint4*)&Vp[(size_t)drow * S_ + kc + 16 + g * 8];
      o[d] = wmma_f16(pf.h, vf.h, o[d]);
    }
  }

  // normalize and store f16 [M, E] at head columns
  #pragma unroll
  for (int r = 0; r < 8; ++r) {
    float inv = 1.0f / li[r];
    int row = b * S_ + q0 + g * 8 + r;
    #pragma unroll
    for (int d = 0; d < 4; ++d) {
      int colc = h * DH_ + d * 16 + lo;
      Out[(size_t)row * E_ + colc] = (_Float16)(o[d][r] * inv);
    }
  }
}

// ---------------- LayerNorm over E=512, one block per row ----------------
__global__ __launch_bounds__(256)
void ln_kernel(const float* __restrict__ X, const float* __restrict__ gam,
               const float* __restrict__ bet, float* __restrict__ Of,
               _Float16* __restrict__ Oh)
{
  __shared__ float red[16];
  const int row = blockIdx.x;
  const int t = threadIdx.x;
  const float* xr = X + (size_t)row * E_;
  float v0 = xr[t], v1 = xr[t + 256];
  float s = v0 + v1, s2 = v0 * v0 + v1 * v1;
  #pragma unroll
  for (int off = 1; off < 32; off <<= 1) {
    s  += __shfl_xor(s,  off, 32);
    s2 += __shfl_xor(s2, off, 32);
  }
  const int wave = t >> 5, lane = t & 31;
  if (lane == 0) { red[wave] = s; red[8 + wave] = s2; }
  __syncthreads();
  if (t < 32) {
    float a  = (lane < 8) ? red[lane]     : 0.f;
    float a2 = (lane < 8) ? red[8 + lane] : 0.f;
    #pragma unroll
    for (int off = 1; off < 8; off <<= 1) {
      a  += __shfl_xor(a,  off, 32);
      a2 += __shfl_xor(a2, off, 32);
    }
    if (lane == 0) { red[0] = a; red[8] = a2; }
  }
  __syncthreads();
  float mu  = red[0] * (1.0f / E_);
  float var = red[8] * (1.0f / E_) - mu * mu;
  float rs  = rsqrtf(var + 1e-5f);
  float y0 = (v0 - mu) * rs * gam[t]       + bet[t];
  float y1 = (v1 - mu) * rs * gam[t + 256] + bet[t + 256];
  if (Of) { Of[(size_t)row * E_ + t] = y0; Of[(size_t)row * E_ + t + 256] = y1; }
  if (Oh) { Oh[(size_t)row * E_ + t] = (_Float16)y0; Oh[(size_t)row * E_ + t + 256] = (_Float16)y1; }
}

// ---------------- launcher ----------------
extern "C" void kernel_launch(void* const* d_in, const int* in_sizes, int n_in,
                              void* d_out, int out_size, void* d_ws, size_t ws_size,
                              hipStream_t stream)
{
  const float* x    = (const float*)d_in[0];
  const float* wqkv = (const float*)d_in[1];
  const float* bqkv = (const float*)d_in[2];
  const float* wout = (const float*)d_in[3];
  const float* bout = (const float*)d_in[4];
  const float* w1   = (const float*)d_in[5];
  const float* b1   = (const float*)d_in[6];
  const float* w2   = (const float*)d_in[7];
  const float* b2   = (const float*)d_in[8];
  const float* ln1g = (const float*)d_in[9];
  const float* ln1b = (const float*)d_in[10];
  const float* ln2g = (const float*)d_in[11];
  const float* ln2b = (const float*)d_in[12];

  if (ws_size < 90177536u) return;  // bump-allocated workspace layout below

  char* ws = (char*)d_ws;
  _Float16* wqkv_h = (_Float16*)(ws + 0);          // 1.5 MB
  _Float16* wout_h = (_Float16*)(ws + 1572864);    // 0.5 MB
  _Float16* w1_h   = (_Float16*)(ws + 2097152);    // 2 MB
  _Float16* w2_h   = (_Float16*)(ws + 4194304);    // 2 MB
  char* rA = ws + 6291456;    // 32 MB: qkv_h, later h_h
  char* rB = ws + 39845888;   // 24 MB: Q,K,Vt, later ln1_f32 + xh2
  char* rC = ws + 65011712;   //  8 MB: xh, later attn_h
  char* rD = ws + 73400320;   // 16 MB: y1, later y2

  _Float16* qkv_h  = (_Float16*)rA;
  _Float16* h_h    = (_Float16*)rA;
  _Float16* Qh     = (_Float16*)rB;
  _Float16* Kh     = (_Float16*)(rB + 8388608);
  _Float16* Vt     = (_Float16*)(rB + 16777216);
  float*    ln1f   = (float*)rB;
  _Float16* xh2    = (_Float16*)(rB + 16777216);
  _Float16* xh     = (_Float16*)rC;
  _Float16* attn_h = (_Float16*)rC;
  float*    y1     = (float*)rD;
  float*    y2     = (float*)rD;

  dim3 blk256(256);

  // precision downcast (one time, deterministic)
  f32_to_f16_kernel<<<(M_ * E_ + 255) / 256,     blk256, 0, stream>>>(x,    xh,     M_ * E_);
  f32_to_f16_kernel<<<(3 * E_ * E_ + 255) / 256, blk256, 0, stream>>>(wqkv, wqkv_h, 3 * E_ * E_);
  f32_to_f16_kernel<<<(E_ * E_ + 255) / 256,     blk256, 0, stream>>>(wout, wout_h, E_ * E_);
  f32_to_f16_kernel<<<(F_ * E_ + 255) / 256,     blk256, 0, stream>>>(w1,   w1_h,   F_ * E_);
  f32_to_f16_kernel<<<(E_ * F_ + 255) / 256,     blk256, 0, stream>>>(w2,   w2_h,   E_ * F_);

  // qkv = x @ in_proj_w^T + b  (f16 out)
  wmma_gemm_kernel<<<dim3(1536 / 128, M_ / 128), blk256, 0, stream>>>(
      xh, wqkv_h, bqkv, nullptr, nullptr, qkv_h, M_, 1536, 512, 0);

  qkv_reshape_kernel<<<(M_ * E_) / 256, blk256, 0, stream>>>(qkv_h, Qh, Kh, Vt);

  attn_kernel<<<dim3(S_ / 16, H_, B_), dim3(32), 0, stream>>>(Qh, Kh, Vt, attn_h);

  // y1 = x + attn @ out_proj_w^T + b  (f32 out)
  wmma_gemm_kernel<<<dim3(512 / 128, M_ / 128), blk256, 0, stream>>>(
      attn_h, wout_h, bout, x, y1, nullptr, M_, 512, 512, 0);

  ln_kernel<<<M_, blk256, 0, stream>>>(y1, ln1g, ln1b, ln1f, xh2);

  // h = relu(ln1 @ w1^T + b1)  (f16 out)
  wmma_gemm_kernel<<<dim3(2048 / 128, M_ / 128), blk256, 0, stream>>>(
      xh2, w1_h, b1, nullptr, nullptr, h_h, M_, 2048, 512, 1);

  // y2 = ln1 + h @ w2^T + b2  (f32 out)
  wmma_gemm_kernel<<<dim3(512 / 128, M_ / 128), blk256, 0, stream>>>(
      h_h, w2_h, b2, ln1f, y2, nullptr, M_, 512, 2048, 0);

  ln_kernel<<<M_, blk256, 0, stream>>>(y2, ln2g, ln2b, (float*)d_out, nullptr);
}
